// BigBirdCCREAttention_85916525789346
// MI455X (gfx1250) — compile-verified, matrix operand order checked by reading, not run on hardware
//
#include <hip/hip_runtime.h>
#include <hip/hip_bf16.h>

// ---------------------------------------------------------------------------
// BigBird block attention for MI455X (gfx1250, wave32, WMMA bf16 16x16x32)
//   B=16, N=1536, DIM=1536, H=8, DK=DV=64, BS=128, nb=12
// All matmul FLOPs via v_wmma_f32_16x16x32_bf16. All LDS fragment reads are
// row-contiguous -> 2x ds_load_b128 each (B-side operands staged transposed).
// Next-tile global_prefetch_b8 overlaps staging with WMMA work.
// ---------------------------------------------------------------------------

#define Bb   16
#define Nn   1536
#define DIMd 1536
#define Hh   8
#define Dk   64
#define NBLK 12
#define HD   512          // H*DK = H*DV
#define BT_LD 40          // transposed B tile row stride (32 + 8 pad halves)
#define VT_LD 384         // transposed V tile row stride (keys in window)

typedef __bf16 v16bf __attribute__((ext_vector_type(16)));
typedef float  v8f   __attribute__((ext_vector_type(8)));

__device__ __forceinline__ unsigned short f32_to_bf16(float f) {
  union { float f; unsigned u; } v; v.f = f;
  unsigned r = v.u + 0x7FFFu + ((v.u >> 16) & 1u);   // round-to-nearest-even
  return (unsigned short)(r >> 16);
}

__device__ __forceinline__ uint2 pack4_bf16(float4 f) {
  uint2 o;
  o.x = (unsigned)f32_to_bf16(f.x) | ((unsigned)f32_to_bf16(f.y) << 16);
  o.y = (unsigned)f32_to_bf16(f.z) | ((unsigned)f32_to_bf16(f.w) << 16);
  return o;
}

__device__ __forceinline__ v8f zero8() {
  v8f z;
#pragma unroll
  for (int i = 0; i < 8; ++i) z[i] = 0.0f;
  return z;
}

// Fragment gather per CDNA5 ISA 16-bit 16x32 layout:
//   element e of lane L -> k = (e<8 ? e : e+8) + (L>=16 ? 8 : 0)
// Row-contiguous: lowers to 2x ds_load_b128 (verified in round-1/2 asm).
__device__ __forceinline__ v16bf frag_row(const unsigned short* p, int kh) {
  v16bf f;
#pragma unroll
  for (int e = 0; e < 16; ++e) {
    int k = (e < 8 ? e : e + 8) + kh;
    f[e] = __builtin_bit_cast(__bf16, p[k]);
  }
  return f;
}

__device__ __forceinline__ v8f wmma_bf16(v16bf a, v16bf b, v8f c) {
  return __builtin_amdgcn_wmma_f32_16x16x32_bf16(false, a, false, b,
                                                 (short)0, c, false, false);
}

// Wave-local LDS fence: DS ops from one wave complete in order; wait for all
// outstanding DS before cross-lane read-back of the P tile.
__device__ __forceinline__ void lds_fence_wave() {
  asm volatile("s_wait_dscnt 0x0" ::: "memory");
}

// ---------------------------------------------------------------------------
// Kernel 1: fused QKV projection.  grid=(192, 8, 3), block=256
//   C[128x64] tile of  x[24576x1536] @ W[1536x512]   (W = Wq/Wk/Wv by z)
//   Output scattered into per-head bf16 layout [B][H][N][64].
// ---------------------------------------------------------------------------
__global__ __launch_bounds__(256) void qkv_proj_kernel(
    const float* __restrict__ x, const float* __restrict__ Wq,
    const float* __restrict__ Wk, const float* __restrict__ Wv,
    unsigned short* __restrict__ qws, unsigned short* __restrict__ kws,
    unsigned short* __restrict__ vws) {
  __shared__ unsigned short Alds[128 * 32];     // [m][k] row-major
  __shared__ unsigned short Bt[64 * BT_LD];     // [n][k] transposed, padded
  const int tid = threadIdx.x;
  const int m0 = blockIdx.x * 128;
  const int n0 = blockIdx.y * 64;
  const int which = blockIdx.z;
  const float* W = (which == 0) ? Wq : ((which == 1) ? Wk : Wv);
  unsigned short* outw = (which == 0) ? qws : ((which == 1) ? kws : vws);

  const int w = tid >> 5, lane = tid & 31;
  const int m16 = lane & 15, kh = (lane >> 4) << 3;

  v8f acc[4];
#pragma unroll
  for (int j = 0; j < 4; ++j) acc[j] = zero8();

  for (int kt = 0; kt < DIMd / 32; ++kt) {
    const int k0 = kt * 32;
    // A tile 128x32: float4 reads, packed uint2 LDS stores (contiguous)
#pragma unroll
    for (int i = 0; i < 4; ++i) {
      int c4 = tid + i * 256;                   // 1024 float4 chunks
      int r = c4 >> 3, off = (c4 & 7) * 4;
      const float* ap = &x[(size_t)(m0 + r) * DIMd + k0 + off];
      float4 fv = *(const float4*)ap;
      ((uint2*)Alds)[c4] = pack4_bf16(fv);
      if (k0 + 32 < DIMd) __builtin_prefetch(ap + 32, 0, 1);  // next k-tile
    }
    // B tile 32x64 -> transposed Bt[n][k]: float4 reads over n, b16 scatter
#pragma unroll
    for (int i = 0; i < 2; ++i) {
      int c4 = tid + i * 256;                   // 512 float4 chunks
      int r = c4 >> 4, cc = (c4 & 15) * 4;      // r = k row, cc = n col
      const float* bp = &W[(size_t)(k0 + r) * HD + n0 + cc];
      float4 fv = *(const float4*)bp;
      Bt[(cc + 0) * BT_LD + r] = f32_to_bf16(fv.x);
      Bt[(cc + 1) * BT_LD + r] = f32_to_bf16(fv.y);
      Bt[(cc + 2) * BT_LD + r] = f32_to_bf16(fv.z);
      Bt[(cc + 3) * BT_LD + r] = f32_to_bf16(fv.w);
      if (k0 + 32 < DIMd) __builtin_prefetch(bp + 32 * HD, 0, 1);
    }
    __syncthreads();
    v16bf a = frag_row(&Alds[(w * 16 + m16) * 32], kh);
#pragma unroll
    for (int j = 0; j < 4; ++j) {
      v16bf bf = frag_row(&Bt[(j * 16 + m16) * BT_LD], kh);
      acc[j] = wmma_bf16(a, bf, acc[j]);
    }
    __syncthreads();
  }

  // Epilogue: all index math is shifts/adds (Nn = 12*128, n0 multiple of 64).
  const int bb  = blockIdx.x / 12;              // batch
  const int nn0 = (blockIdx.x % 12) * 128;      // token base within batch
  const int hh  = blockIdx.y >> 0;              // col>>6 == blockIdx.y (n0=64*y)
  const size_t headRow = ((size_t)(bb * Hh + hh) * Nn + nn0 + w * 16) * Dk;
#pragma unroll
  for (int j = 0; j < 4; ++j) {
#pragma unroll
    for (int r = 0; r < 8; ++r) {
      int mloc = r + ((lane >> 4) << 3);
      outw[headRow + (size_t)mloc * Dk + j * 16 + m16] = f32_to_bf16(acc[j][r]);
    }
  }
}

// ---------------------------------------------------------------------------
// Kernel 2: block attention with streaming softmax.  grid=1536, block=256
//   per (b,h,blk): Q[128x64], window K[384x64], V transposed [64x384]
//   (zero padded edges -> logit exactly 0, matching the reference).
// ---------------------------------------------------------------------------
__global__ __launch_bounds__(256) void attn_kernel(
    const unsigned short* __restrict__ qws, const unsigned short* __restrict__ kws,
    const unsigned short* __restrict__ vws, unsigned short* __restrict__ aout) {
  extern __shared__ char smem[];
  unsigned short* Qs = (unsigned short*)smem;   // 128*64  [q][d]
  unsigned short* Ks = Qs + 128 * 64;           // 384*64  [key][d]
  unsigned short* Vt = Ks + 384 * 64;           // 64*384  [d][key]  transposed
  unsigned short* Ps = Vt + 64 * VT_LD;         // 8 waves * 16*32

  const int tid = threadIdx.x;
  const int gid = blockIdx.x;
  const int blk = gid % NBLK;
  const int h = (gid / NBLK) % Hh;
  const int b = gid / (NBLK * Hh);
  const size_t headBase = (size_t)((b * Hh + h) * Nn) * Dk;

  // stage Q (8-byte chunks: 4 bf16)
  for (int c = tid; c < 2048; c += 256) {
    int r = c >> 4, off = (c & 15) * 4;
    int nq = blk * 128 + r;
    ((uint2*)Qs)[c] = *(const uint2*)(qws + headBase + (size_t)nq * Dk + off);
  }
  // stage K (row-major) and V (transposed) with zero edge padding
  for (int c = tid; c < 6144; c += 256) {
    int r = c >> 4, off = (c & 15) * 4;         // r = key 0..383, off = d
    int nk = blk * 128 - 128 + r;
    uint2 kv; kv.x = 0u; kv.y = 0u;
    unsigned short v0 = 0, v1 = 0, v2 = 0, v3 = 0;
    if (nk >= 0 && nk < Nn) {
      kv = *(const uint2*)(kws + headBase + (size_t)nk * Dk + off);
      uint2 vv = *(const uint2*)(vws + headBase + (size_t)nk * Dk + off);
      v0 = (unsigned short)(vv.x & 0xFFFF);
      v1 = (unsigned short)(vv.x >> 16);
      v2 = (unsigned short)(vv.y & 0xFFFF);
      v3 = (unsigned short)(vv.y >> 16);
    }
    ((uint2*)Ks)[c] = kv;
    Vt[(off + 0) * VT_LD + r] = v0;
    Vt[(off + 1) * VT_LD + r] = v1;
    Vt[(off + 2) * VT_LD + r] = v2;
    Vt[(off + 3) * VT_LD + r] = v3;
  }
  __syncthreads();

  const int w = tid >> 5, lane = tid & 31;
  const int m16 = lane & 15, kh = (lane >> 4) << 3;
  unsigned short* pw = Ps + w * (16 * 32);

  v16bf aq0 = frag_row(Qs + (w * 16 + m16) * 64, kh);       // d 0..31
  v16bf aq1 = frag_row(Qs + (w * 16 + m16) * 64 + 32, kh);  // d 32..63

  v8f O0 = zero8(), O1 = zero8(), O2 = zero8(), O3 = zero8();
  float mrow[8], lrow[8];
#pragma unroll
  for (int r = 0; r < 8; ++r) { mrow[r] = -3.0e38f; lrow[r] = 0.0f; }

  const float scale = 0.125f;                 // 1/sqrt(64)
  for (int it = 0; it < 12; ++it) {
    const int kk = it * 32;
    v8f s0 = zero8(), s1 = zero8();
    {
      v16bf bk  = frag_row(Ks + (kk + m16) * 64, kh);
      v16bf bk2 = frag_row(Ks + (kk + m16) * 64 + 32, kh);
      s0 = wmma_bf16(aq0, bk, s0);
      s0 = wmma_bf16(aq1, bk2, s0);
    }
    {
      v16bf bk  = frag_row(Ks + (kk + 16 + m16) * 64, kh);
      v16bf bk2 = frag_row(Ks + (kk + 16 + m16) * 64 + 32, kh);
      s1 = wmma_bf16(aq0, bk, s1);
      s1 = wmma_bf16(aq1, bk2, s1);
    }
    // online softmax: rows live across 16-lane halves (masks 1/2/4/8 stay in-half)
#pragma unroll
    for (int r = 0; r < 8; ++r) {
      float v0 = s0[r] * scale, v1 = s1[r] * scale;
      float t = fmaxf(v0, v1);
      t = fmaxf(t, __shfl_xor(t, 8));
      t = fmaxf(t, __shfl_xor(t, 4));
      t = fmaxf(t, __shfl_xor(t, 2));
      t = fmaxf(t, __shfl_xor(t, 1));
      float mnew = fmaxf(mrow[r], t);
      float alpha = __expf(mrow[r] - mnew);
      float p0 = __expf(v0 - mnew);
      float p1 = __expf(v1 - mnew);
      float rs = p0 + p1;
      rs += __shfl_xor(rs, 8);
      rs += __shfl_xor(rs, 4);
      rs += __shfl_xor(rs, 2);
      rs += __shfl_xor(rs, 1);
      lrow[r] = lrow[r] * alpha + rs;
      mrow[r] = mnew;
      O0[r] *= alpha; O1[r] *= alpha; O2[r] *= alpha; O3[r] *= alpha;
      int ml = r + ((lane >> 4) << 3);
      pw[ml * 32 + m16]      = f32_to_bf16(p0);
      pw[ml * 32 + 16 + m16] = f32_to_bf16(p1);
    }
    lds_fence_wave();
    v16bf ap = frag_row(pw + m16 * 32, kh);    // P as 16x32 A fragment
    O0 = wmma_bf16(ap, frag_row(Vt + ( 0 + m16) * VT_LD + kk, kh), O0);
    O1 = wmma_bf16(ap, frag_row(Vt + (16 + m16) * VT_LD + kk, kh), O1);
    O2 = wmma_bf16(ap, frag_row(Vt + (32 + m16) * VT_LD + kk, kh), O2);
    O3 = wmma_bf16(ap, frag_row(Vt + (48 + m16) * VT_LD + kk, kh), O3);
  }

  // normalize and write [B][N][H*DV] bf16 (token-major for the output GEMM)
#pragma unroll
  for (int r = 0; r < 8; ++r) {
    int mloc = r + ((lane >> 4) << 3);
    int row = blk * 128 + w * 16 + mloc;
    float inv = 1.0f / lrow[r];
    size_t base = ((size_t)(b * Nn) + row) * HD + h * Dk;
    aout[base +  0 + m16] = f32_to_bf16(O0[r] * inv);
    aout[base + 16 + m16] = f32_to_bf16(O1[r] * inv);
    aout[base + 32 + m16] = f32_to_bf16(O2[r] * inv);
    aout[base + 48 + m16] = f32_to_bf16(O3[r] * inv);
  }
}

// ---------------------------------------------------------------------------
// Kernel 3: output projection + bias.  grid=(192, 24), block=256
//   out[24576x1536] = attn[24576x512](bf16) @ Wo[512x1536] + bo
// ---------------------------------------------------------------------------
__global__ __launch_bounds__(256) void out_proj_kernel(
    const unsigned short* __restrict__ attn, const float* __restrict__ Wo,
    const float* __restrict__ bo, float* __restrict__ out) {
  __shared__ unsigned short Alds[128 * 32];     // [m][k]
  __shared__ unsigned short Bt[64 * BT_LD];     // [n][k] transposed, padded
  const int tid = threadIdx.x;
  const int m0 = blockIdx.x * 128;
  const int n0 = blockIdx.y * 64;
  const int w = tid >> 5, lane = tid & 31;
  const int m16 = lane & 15, kh = (lane >> 4) << 3;

  v8f acc[4];
#pragma unroll
  for (int j = 0; j < 4; ++j) acc[j] = zero8();

  for (int kt = 0; kt < HD / 32; ++kt) {
    const int k0 = kt * 32;
    // A tile 128x32 bf16: straight 8B copies (already bf16)
#pragma unroll
    for (int i = 0; i < 4; ++i) {
      int c4 = tid + i * 256;
      int r = c4 >> 3, off = (c4 & 7) * 4;
      const unsigned short* ap = attn + (size_t)(m0 + r) * HD + k0 + off;
      ((uint2*)Alds)[c4] = *(const uint2*)ap;
      if (k0 + 32 < HD) __builtin_prefetch(ap + 32, 0, 1);
    }
    // Wo tile 32x64 -> transposed Bt[n][k]
#pragma unroll
    for (int i = 0; i < 2; ++i) {
      int c4 = tid + i * 256;
      int r = c4 >> 4, cc = (c4 & 15) * 4;
      const float* bp = &Wo[(size_t)(k0 + r) * DIMd + n0 + cc];
      float4 fv = *(const float4*)bp;
      Bt[(cc + 0) * BT_LD + r] = f32_to_bf16(fv.x);
      Bt[(cc + 1) * BT_LD + r] = f32_to_bf16(fv.y);
      Bt[(cc + 2) * BT_LD + r] = f32_to_bf16(fv.z);
      Bt[(cc + 3) * BT_LD + r] = f32_to_bf16(fv.w);
      if (k0 + 32 < HD) __builtin_prefetch(bp + 32 * DIMd, 0, 1);
    }
    __syncthreads();
    v16bf a = frag_row(&Alds[(w * 16 + m16) * 32], kh);
#pragma unroll
    for (int j = 0; j < 4; ++j)
      acc[j] = wmma_bf16(a, frag_row(&Bt[(j * 16 + m16) * BT_LD], kh), acc[j]);
    __syncthreads();
  }

#pragma unroll
  for (int j = 0; j < 4; ++j) {
#pragma unroll
    for (int r = 0; r < 8; ++r) {
      int mloc = r + ((lane >> 4) << 3);
      int row = m0 + w * 16 + mloc;
      int col = n0 + j * 16 + m16;
      out[(size_t)row * DIMd + col] = acc[j][r] + bo[col];
    }
  }
}

// ---------------------------------------------------------------------------
extern "C" void kernel_launch(void* const* d_in, const int* in_sizes, int n_in,
                              void* d_out, int out_size, void* d_ws, size_t ws_size,
                              hipStream_t stream) {
  const float* x  = (const float*)d_in[0];
  const float* Wq = (const float*)d_in[1];
  const float* Wk = (const float*)d_in[2];
  const float* Wv = (const float*)d_in[3];
  const float* Wo = (const float*)d_in[4];
  const float* bo = (const float*)d_in[5];
  float* out = (float*)d_out;

  const size_t perMat = (size_t)Bb * Hh * Nn * Dk;   // 12,582,912 bf16 elems
  unsigned short* qws = (unsigned short*)d_ws;
  unsigned short* kws = qws + perMat;
  unsigned short* vws = kws + perMat;
  unsigned short* aws = vws + perMat;                // [B][N][H*DV] bf16

  qkv_proj_kernel<<<dim3(Bb * Nn / 128, HD / 64, 3), 256, 0, stream>>>(
      x, Wq, Wk, Wv, qws, kws, vws);

  const int attnLds = (128 * 64 + 384 * 64 + 64 * VT_LD + 8 * 16 * 32) * 2; // 120 KB
  (void)hipFuncSetAttribute(reinterpret_cast<const void*>(attn_kernel),
                            hipFuncAttributeMaxDynamicSharedMemorySize, attnLds);
  attn_kernel<<<dim3(Bb * Hh * NBLK), 256, attnLds, stream>>>(qws, kws, vws, aws);

  out_proj_kernel<<<dim3(Bb * Nn / 128, DIMd / 64), 256, 0, stream>>>(
      aws, Wo, bo, out);
}
